// CausalSelfAttention_80444737454177
// MI455X (gfx1250) — matrix operand order repaired and checked by reading, not compiled, not run on hardware
//
#include <hip/hip_runtime.h>
#include <math.h>

#define B_    2
#define T_    2048
#define C_    1024
#define NH_   16
#define NKV_  4
#define HD_   64
#define WIN_  1024
#define EPS_  1.1920929e-07f

typedef __bf16 bf16;
typedef __attribute__((ext_vector_type(16))) __bf16 v16bf;
typedef __attribute__((ext_vector_type(8)))  float  v8f;

union FragA { uint4 u[2]; v16bf v; };
union VBuf  { uint4 u[8]; bf16 h[64]; };

static __device__ inline v8f wmma_bf16(v16bf a, v16bf b, v8f c) {
  return __builtin_amdgcn_wmma_f32_16x16x32_bf16(false, a, false, b, (short)0, c, false, false);
}

// CDNA5 async global->LDS copy, 16B per lane, tracked by ASYNCcnt.
// vdst = 32-bit LDS byte address (low 32 bits of the generic pointer),
// vaddr = 64-bit global address, GV mode (saddr = off).
static __device__ inline void async_ld16(const bf16* g, bf16* l) {
  unsigned loff = (unsigned)(size_t)(void*)l;
  unsigned long long ga = (unsigned long long)g;
  asm volatile("global_load_async_to_lds_b128 %0, %1, off"
               :: "v"(loff), "v"(ga) : "memory");
}

// ---------------------------------------------------------------------------
// f32 -> bf16 elementwise convert (weights)
// ---------------------------------------------------------------------------
__global__ void f2bf_kernel(const float* __restrict__ src, bf16* __restrict__ dst, int n) {
  int i = blockIdx.x * blockDim.x + threadIdx.x;
  if (i < n) dst[i] = (bf16)src[i];
}

// ---------------------------------------------------------------------------
// prep: xs = x + rmsnorm(x0); also bf16 copy of x. One block per token row.
// ---------------------------------------------------------------------------
__global__ __launch_bounds__(256)
void prep_kernel(const float* __restrict__ x, const float* __restrict__ x0,
                 bf16* __restrict__ xsb, bf16* __restrict__ xb) {
  __shared__ float red[8];
  int r = blockIdx.x;
  int tid = threadIdx.x;
  int wave = tid >> 5, lane = tid & 31;
  float v0[4], vx[4];
  float local = 0.f;
#pragma unroll
  for (int j = 0; j < 4; j++) {
    int cidx = tid + j * 256;
    v0[j] = x0[(size_t)r * C_ + cidx];
    vx[j] = x[(size_t)r * C_ + cidx];
    local += v0[j] * v0[j];
  }
#pragma unroll
  for (int d = 16; d >= 1; d >>= 1) local += __shfl_xor(local, d);
  if (lane == 0) red[wave] = local;
  __syncthreads();
  float sum = 0.f;
#pragma unroll
  for (int w = 0; w < 8; w++) sum += red[w];
  float rms = rsqrtf(sum * (1.0f / C_) + EPS_);
#pragma unroll
  for (int j = 0; j < 4; j++) {
    int cidx = tid + j * 256;
    xsb[(size_t)r * C_ + cidx] = (bf16)(vx[j] + v0[j] * rms);
    xb[(size_t)r * C_ + cidx]  = (bf16)vx[j];
  }
}

// ---------------------------------------------------------------------------
// WMMA GEMM: out[m,n] = scale * sum_k A[m,k] * W[n,k]
// block = 256 threads (8 waves); block tile M=256, N=64; wave tile 32x64.
// K-step 32, double-buffered LDS staging via global_load_async_to_lds_b128.
// Per iteration/wave: 12 ds_load_b128 feeding 8 back-to-back WMMAs.
// ---------------------------------------------------------------------------
__global__ __launch_bounds__(256)
void gemm_bf16(const bf16* __restrict__ A, const bf16* __restrict__ W,
               float* __restrict__ out, int M, int N, int K, float scale) {
  __shared__ __align__(16) bf16 Alds[2][256 * 32];
  __shared__ __align__(16) bf16 Wlds[2][64 * 32];
  int tid = threadIdx.x;
  int wave = tid >> 5, lane = tid & 31;
  int col = lane & 15, half = lane >> 4;
  int m0 = blockIdx.y * 256, n0 = blockIdx.x * 64;

  // static chunk split: this thread stages A rows (r, r+64, r+128, r+192)
  // and W row r, 16B each -- 5 async ops per tile, zero divergence.
  int srow = tid >> 2, spart = tid & 3;

  v8f acc0[4] = {}, acc1[4] = {};
  int ntiles = K >> 5;

  // prologue: stage tile 0 into buffer 0
#pragma unroll
  for (int rr = 0; rr < 4; rr++) {
    const bf16* ga = A + (size_t)(m0 + rr * 64 + srow) * K + spart * 8;
    async_ld16(ga, &Alds[0][(rr * 64 + srow) * 32 + spart * 8]);
  }
  async_ld16(W + (size_t)(n0 + srow) * K + spart * 8,
             &Wlds[0][srow * 32 + spart * 8]);

  for (int it = 0; it < ntiles; ++it) {
    int cur = it & 1;
    if (it + 1 < ntiles) {
      int k1 = (it + 1) << 5;
#pragma unroll
      for (int rr = 0; rr < 4; rr++) {
        const bf16* ga = A + (size_t)(m0 + rr * 64 + srow) * K + k1 + spart * 8;
        async_ld16(ga, &Alds[cur ^ 1][(rr * 64 + srow) * 32 + spart * 8]);
      }
      async_ld16(W + (size_t)(n0 + srow) * K + k1 + spart * 8,
                 &Wlds[cur ^ 1][srow * 32 + spart * 8]);
      // async loads complete in order: <=5 outstanding => tile `it` is in LDS
      asm volatile("s_wait_asynccnt 0x5" ::: "memory");
    } else {
      asm volatile("s_wait_asynccnt 0x0" ::: "memory");
    }
    __syncthreads();

    // load all fragments first, then burst the WMMAs
    FragA af0, af1, wf[4];
    const bf16* ap0 = &Alds[cur][(wave * 32 + col) * 32];
    const bf16* ap1 = &Alds[cur][(wave * 32 + 16 + col) * 32];
    af0.u[0] = *(const uint4*)(ap0 + half * 8);
    af0.u[1] = *(const uint4*)(ap0 + 16 + half * 8);
    af1.u[0] = *(const uint4*)(ap1 + half * 8);
    af1.u[1] = *(const uint4*)(ap1 + 16 + half * 8);
#pragma unroll
    for (int nt = 0; nt < 4; nt++) {
      const bf16* wp = &Wlds[cur][(nt * 16 + col) * 32 + half * 8];
      wf[nt].u[0] = *(const uint4*)(wp);
      wf[nt].u[1] = *(const uint4*)(wp + 16);
    }
#pragma unroll
    for (int nt = 0; nt < 4; nt++) acc0[nt] = wmma_bf16(af0.v, wf[nt].v, acc0[nt]);
#pragma unroll
    for (int nt = 0; nt < 4; nt++) acc1[nt] = wmma_bf16(af1.v, wf[nt].v, acc1[nt]);
    __syncthreads();
  }

#pragma unroll
  for (int nt = 0; nt < 4; nt++)
#pragma unroll
    for (int v = 0; v < 8; v++) {
      int n = n0 + nt * 16 + col;
      int ma = m0 + wave * 32 + v + 8 * half;
      int mb = ma + 16;
      out[(size_t)ma * N + n] = acc0[nt][v] * scale;
      out[(size_t)mb * N + n] = acc1[nt][v] * scale;
    }
}

// ---------------------------------------------------------------------------
// RoPE + rmsnorm on q (x1.2 * 1/sqrt(HD) folded in) and k (x1.2); v combine.
// One block per token; one wave per head task.
// ---------------------------------------------------------------------------
__global__ __launch_bounds__(256)
void rope_kernel(const float* __restrict__ qraw, const float* __restrict__ kraw,
                 const float* __restrict__ vraw, const float* __restrict__ ve,
                 const float* __restrict__ cosp, const float* __restrict__ sinp,
                 bf16* __restrict__ qb, bf16* __restrict__ kb, bf16* __restrict__ vb) {
  int r = blockIdx.x;
  int t = r & (T_ - 1);
  int tid = threadIdx.x, wave = tid >> 5, lane = tid & 31;
  float c = cosp[(size_t)t * (HD_ / 2) + lane];
  float s = sinp[(size_t)t * (HD_ / 2) + lane];

  for (int task = wave; task < NH_ + NKV_; task += 8) {
    const float* base;
    bf16* ob;
    float sc;
    if (task < NH_) {
      base = qraw + (size_t)r * (NH_ * HD_) + task * HD_;
      ob   = qb   + (size_t)r * (NH_ * HD_) + task * HD_;
      sc   = 1.2f * 0.125f;   // 1.2 * rsqrt(HD)
    } else {
      int hh = task - NH_;
      base = kraw + (size_t)r * (NKV_ * HD_) + hh * HD_;
      ob   = kb   + (size_t)r * (NKV_ * HD_) + hh * HD_;
      sc   = 1.2f;
    }
    float x1 = base[lane], x2 = base[lane + 32];
    float o1 = x1 * c + x2 * s;
    float o2 = -x1 * s + x2 * c;
    float ss = o1 * o1 + o2 * o2;
#pragma unroll
    for (int d = 16; d >= 1; d >>= 1) ss += __shfl_xor(ss, d);
    float rr = rsqrtf(ss * (1.0f / HD_) + EPS_) * sc;
    ob[lane]      = (bf16)(o1 * rr);
    ob[lane + 32] = (bf16)(o2 * rr);
  }
  // v = 0.5*(xs@Wv.T) + 0.5*ve   (first term already scaled in GEMM)
  size_t vi = (size_t)r * (NKV_ * HD_) + tid;
  vb[vi] = (bf16)(vraw[vi] + 0.5f * ve[vi]);
}

// ---------------------------------------------------------------------------
// Windowed-causal flash attention, GQA 16->4 heads, WMMA QK^T and P*V.
// Block = 4 waves (128 thr); each wave owns a 16-query tile independently.
// ---------------------------------------------------------------------------
__global__ __launch_bounds__(128)
void attn_kernel(const bf16* __restrict__ qbuf, const bf16* __restrict__ kbuf,
                 const bf16* __restrict__ vbuf, bf16* __restrict__ ybuf) {
  __shared__ __align__(16) bf16 Plds[4][16 * 32];
  __shared__ __align__(16) bf16 VtLds[4][64 * 32];

  int wave = threadIdx.x >> 5, lane = threadIdx.x & 31;
  int col = lane & 15, half = lane >> 4;
  int h = blockIdx.y, b = blockIdx.z;
  int hk = h >> 2;                               // GQA: kv head = h // 4
  int qt = blockIdx.x * 64 + wave * 16;

  // Q fragments (16x64 split into two 16x32 A fragments), kept in registers
  int qrow = qt + col;
  const bf16* qp = qbuf + (((size_t)(b * T_) + qrow) * NH_ + h) * HD_;
  FragA qa0, qa1;
  qa0.u[0] = *(const uint4*)(qp + half * 8);
  qa0.u[1] = *(const uint4*)(qp + 16 + half * 8);
  qa1.u[0] = *(const uint4*)(qp + 32 + half * 8);
  qa1.u[1] = *(const uint4*)(qp + 48 + half * 8);

  v8f o[4] = {};
  float mrun[8], lrun[8];
#pragma unroll
  for (int v = 0; v < 8; v++) { mrun[v] = -1e30f; lrun[v] = 0.f; }

  bf16* P  = Plds[wave];
  bf16* VT = VtLds[wave];

  int jmin = qt - WIN_;
  if (jmin < 0) jmin = 0;
  int kt0 = (jmin >> 5) << 5;

  for (int kt = kt0; kt <= qt + 15; kt += 32) {
    // --- S = Q K^T for this 32-key tile (two 16-key halves) ---
    int j0 = kt + col;       if (j0 > T_ - 1) j0 = T_ - 1;
    int j1 = kt + 16 + col;  if (j1 > T_ - 1) j1 = T_ - 1;
    const bf16* kp0 = kbuf + (((size_t)(b * T_) + j0) * NKV_ + hk) * HD_;
    const bf16* kp1 = kbuf + (((size_t)(b * T_) + j1) * NKV_ + hk) * HD_;
    FragA kf00, kf01, kf10, kf11;
    kf00.u[0] = *(const uint4*)(kp0 + half * 8);
    kf00.u[1] = *(const uint4*)(kp0 + 16 + half * 8);
    kf01.u[0] = *(const uint4*)(kp0 + 32 + half * 8);
    kf01.u[1] = *(const uint4*)(kp0 + 48 + half * 8);
    kf10.u[0] = *(const uint4*)(kp1 + half * 8);
    kf10.u[1] = *(const uint4*)(kp1 + 16 + half * 8);
    kf11.u[0] = *(const uint4*)(kp1 + 32 + half * 8);
    kf11.u[1] = *(const uint4*)(kp1 + 48 + half * 8);

    v8f s0 = {}, s1 = {};
    s0 = wmma_bf16(qa0.v, kf00.v, s0);
    s0 = wmma_bf16(qa1.v, kf01.v, s0);
    s1 = wmma_bf16(qa0.v, kf10.v, s1);
    s1 = wmma_bf16(qa1.v, kf11.v, s1);

    bool needMask = (kt + 31 > qt) || (qt + 15 - kt > WIN_);

    // --- online softmax per row; P -> LDS (bf16) ---
#pragma unroll
    for (int v = 0; v < 8; v++) {
      int i = qt + v + 8 * half;
      float sv0 = s0[v], sv1 = s1[v];
      if (needMask) {
        int ja = kt + col, jb2 = kt + 16 + col;
        if (ja > i || i - ja > WIN_) sv0 = -1e30f;
        if (jb2 > i || i - jb2 > WIN_) sv1 = -1e30f;
      }
      float mx = fmaxf(sv0, sv1);
      mx = fmaxf(mx, __shfl_xor(mx, 1));
      mx = fmaxf(mx, __shfl_xor(mx, 2));
      mx = fmaxf(mx, __shfl_xor(mx, 4));
      mx = fmaxf(mx, __shfl_xor(mx, 8));
      float mnew  = fmaxf(mrun[v], mx);
      float alpha = __expf(mrun[v] - mnew);
      mrun[v] = mnew;
      float p0 = __expf(sv0 - mnew);
      float p1 = __expf(sv1 - mnew);
      float rs = p0 + p1;
      rs += __shfl_xor(rs, 1);
      rs += __shfl_xor(rs, 2);
      rs += __shfl_xor(rs, 4);
      rs += __shfl_xor(rs, 8);
      lrun[v] = lrun[v] * alpha + rs;
      o[0][v] *= alpha; o[1][v] *= alpha; o[2][v] *= alpha; o[3][v] *= alpha;
      int prow = v + 8 * half;
      P[prow * 32 + col]      = (bf16)p0;
      P[prow * 32 + 16 + col] = (bf16)p1;
    }

    // --- stage V^T (32 keys x 64 dims) in this wave's LDS slice ---
    {
      int j = kt + lane;
      if (j > T_ - 1) j = T_ - 1;
      const bf16* vp = vbuf + (((size_t)(b * T_) + j) * NKV_ + hk) * HD_;
      VBuf vv;
#pragma unroll
      for (int c2 = 0; c2 < 8; c2++) vv.u[c2] = *(const uint4*)(vp + c2 * 8);
#pragma unroll
      for (int d = 0; d < 64; d++) VT[d * 32 + lane] = vv.h[d];
    }

    // --- O += P * V via WMMA (all fragments loaded, then WMMA burst) ---
    FragA pa, vf[4];
    pa.u[0] = *(const uint4*)&P[col * 32 + half * 8];
    pa.u[1] = *(const uint4*)&P[col * 32 + 16 + half * 8];
#pragma unroll
    for (int nt = 0; nt < 4; nt++) {
      const bf16* vtp = &VT[(nt * 16 + col) * 32 + half * 8];
      vf[nt].u[0] = *(const uint4*)(vtp);
      vf[nt].u[1] = *(const uint4*)(vtp + 16);
    }
#pragma unroll
    for (int nt = 0; nt < 4; nt++) o[nt] = wmma_bf16(pa.v, vf[nt].v, o[nt]);
  }

  // --- normalize and store (b, i, h*64 + d) ---
#pragma unroll
  for (int nt = 0; nt < 4; nt++)
#pragma unroll
    for (int v = 0; v < 8; v++) {
      int i = qt + v + 8 * half;
      float val = o[nt][v] / lrun[v];
      ybuf[((size_t)(b * T_) + i) * C_ + h * HD_ + nt * 16 + col] = (bf16)val;
    }
}

// ---------------------------------------------------------------------------
extern "C" void kernel_launch(void* const* d_in, const int* in_sizes, int n_in,
                              void* d_out, int out_size, void* d_ws, size_t ws_size,
                              hipStream_t stream) {
  const float* x    = (const float*)d_in[0];
  const float* x0   = (const float*)d_in[1];
  const float* ve   = (const float*)d_in[2];
  const float* cosp = (const float*)d_in[3];
  const float* sinp = (const float*)d_in[4];
  const float* Wq   = (const float*)d_in[5];
  const float* Wk   = (const float*)d_in[6];
  const float* Wv   = (const float*)d_in[7];
  const float* Wo   = (const float*)d_in[8];
  float* out = (float*)d_out;

  const int BT = B_ * T_;
  char* p = (char*)d_ws;
  auto alloc = [&](size_t bytes) -> char* {
    char* r = p;
    p += (bytes + 255) & ~(size_t)255;
    return r;
  };
  bf16* xsb   = (bf16*)alloc((size_t)BT * C_ * 2);
  bf16* xb    = (bf16*)alloc((size_t)BT * C_ * 2);
  bf16* wqb   = (bf16*)alloc((size_t)C_ * C_ * 2);
  bf16* wkb   = (bf16*)alloc((size_t)(NKV_ * HD_) * C_ * 2);
  bf16* wvb   = (bf16*)alloc((size_t)(NKV_ * HD_) * C_ * 2);
  bf16* wob   = (bf16*)alloc((size_t)C_ * C_ * 2);
  float* qraw = (float*)alloc((size_t)BT * (NH_ * HD_) * 4);
  float* kraw = (float*)alloc((size_t)BT * (NKV_ * HD_) * 4);
  float* vraw = (float*)alloc((size_t)BT * (NKV_ * HD_) * 4);
  bf16* qb    = (bf16*)alloc((size_t)BT * (NH_ * HD_) * 2);
  bf16* kb    = (bf16*)alloc((size_t)BT * (NKV_ * HD_) * 2);
  bf16* vb    = (bf16*)alloc((size_t)BT * (NKV_ * HD_) * 2);
  bf16* yatt  = (bf16*)alloc((size_t)BT * C_ * 2);

  // weights -> bf16
  f2bf_kernel<<<(C_ * C_ + 255) / 256, 256, 0, stream>>>(Wq, wqb, C_ * C_);
  f2bf_kernel<<<(NKV_ * HD_ * C_ + 255) / 256, 256, 0, stream>>>(Wk, wkb, NKV_ * HD_ * C_);
  f2bf_kernel<<<(NKV_ * HD_ * C_ + 255) / 256, 256, 0, stream>>>(Wv, wvb, NKV_ * HD_ * C_);
  f2bf_kernel<<<(C_ * C_ + 255) / 256, 256, 0, stream>>>(Wo, wob, C_ * C_);

  // xs = x + rmsnorm(x0)
  prep_kernel<<<BT, 256, 0, stream>>>(x, x0, xsb, xb);

  // fused-stream QKV projections (scale 0.5) + v_init = x @ Wv.T (to d_out)
  gemm_bf16<<<dim3((NH_ * HD_) / 64, BT / 256), 256, 0, stream>>>(
      xsb, wqb, qraw, BT, NH_ * HD_, C_, 0.5f);
  gemm_bf16<<<dim3((NKV_ * HD_) / 64, BT / 256), 256, 0, stream>>>(
      xsb, wkb, kraw, BT, NKV_ * HD_, C_, 0.5f);
  gemm_bf16<<<dim3((NKV_ * HD_) / 64, BT / 256), 256, 0, stream>>>(
      xsb, wvb, vraw, BT, NKV_ * HD_, C_, 0.5f);
  gemm_bf16<<<dim3((NKV_ * HD_) / 64, BT / 256), 256, 0, stream>>>(
      xb, wvb, out + (size_t)BT * C_, BT, NKV_ * HD_, C_, 1.0f);

  // RoPE + rmsnorm + v combine
  rope_kernel<<<BT, 256, 0, stream>>>(qraw, kraw, vraw, ve, cosp, sinp, qb, kb, vb);

  // windowed causal flash attention
  attn_kernel<<<dim3(T_ / 64, NH_, B_), 128, 0, stream>>>(qb, kb, vb, yatt);

  // output projection -> d_out
  gemm_bf16<<<dim3(C_ / 64, BT / 256), 256, 0, stream>>>(yatt, wob, out, BT, C_, C_, 1.0f);
}
